// GCN_FCNet_25692494365029
// MI455X (gfx1250) — matrix operand order repaired and checked by reading, not compiled, not run on hardware
//
#include <hip/hip_runtime.h>

// ---- problem dims ----
#define NV_    117000
#define NE_    468000
#define EMB_   300
#define EMBP_N 304    // 19 * 16  (N padding for GEMM tiles)
#define EMBP_K 320    // 10 * 32  (K padding for WMMA K=32 steps)
#define INDIM_ 768
#define BATCH_ 4096
#define FCK_   1068   // EMB + IN_DIM
#define FCKP_  1088   // 34 * 32
#define HID_   500
#define HIDP_  512    // 32 * 16 and 16 * 32
#define NLAB_  2

typedef __attribute__((ext_vector_type(16))) _Float16     v16h;
typedef __attribute__((ext_vector_type(8)))  float        v8f;
typedef __attribute__((ext_vector_type(8)))  unsigned int v8u;
typedef unsigned int uint32;

__device__ __forceinline__ float sigmoidf_(float x) {
    return 1.0f / (1.0f + __expf(-x));
}

// low 32 bits of a generic pointer to LDS = LDS byte offset (flat aperture scheme)
__device__ __forceinline__ uint32 lds_off_(const void* p) {
    return (uint32)(uintptr_t)p;
}

// GLOBAL_LOAD_ASYNC_TO_LDS_B128: VDST = LDS byte-offset VGPR, VADDR = 64-bit address
__device__ __forceinline__ void async_b128_(uint32 lds, const void* g) {
    asm volatile("global_load_async_to_lds_b128 %0, %1, off" :: "v"(lds), "v"(g) : "memory");
}

template <int N>
__device__ __forceinline__ void wait_async_() {
    asm volatile("s_wait_asynccnt %0" :: "i"(N) : "memory");
}

// ---------------- utility kernels ----------------

__global__ void k_zero_f32(float* __restrict__ p, int n) {
    int i = blockIdx.x * blockDim.x + threadIdx.x;
    if (i < n) p[i] = 0.0f;
}

__global__ void k_degree(const int* __restrict__ er, float* __restrict__ deg, int ne) {
    int e = blockIdx.x * blockDim.x + threadIdx.x;
    if (e < ne) atomicAdd(&deg[er[e]], 1.0f);
}

__global__ void k_dinv(const float* __restrict__ deg, float* __restrict__ dinv, int nv) {
    int v = blockIdx.x * blockDim.x + threadIdx.x;
    if (v < nv) {
        float d = deg[v];
        dinv[v] = (d > 0.0f) ? rsqrtf(d) : 0.0f;
    }
}

// H0 (f32 [NV][300]) -> Hf16 ([NV][EMBP_K]), zero K-pad
__global__ void k_cvt_h0(const float* __restrict__ H0, _Float16* __restrict__ Hf, int total) {
    int i = blockIdx.x * blockDim.x + threadIdx.x;
    if (i >= total) return;
    int f = i % EMBP_K;
    int v = i / EMBP_K;
    Hf[i] = (f < EMB_) ? (_Float16)H0[(size_t)v * EMB_ + f] : (_Float16)0.0f;
}

// W (f32 [K][N]) -> Wt (f16 [Npad][Kpad]) transposed + zero pad
__global__ void k_cvt_w(const float* __restrict__ W, _Float16* __restrict__ Wt,
                        int K, int N, int Kpad, int Npad) {
    int i = blockIdx.x * blockDim.x + threadIdx.x;
    int total = Npad * Kpad;
    if (i >= total) return;
    int k = i % Kpad;
    int n = i / Kpad;
    Wt[i] = (k < K && n < N) ? (_Float16)W[(size_t)k * N + n] : (_Float16)0.0f;
}

// ---------------- WMMA GEMM ----------------
// C[M][outStride] = A[M][Kpad] * Bt[Npad][Kpad]^T  (+bias, sigmoid optional)
// Block = 128 threads (4 waves) -> 64x64 macro tile. Wave w owns m-tile mg*4+w and
// 4 n-tiles (4 f32 accumulators; A fragment reused). B slab (64 rows x 32 K halves
// = 4KB) is staged through LDS with async b128 copies, double buffered.
__global__ void __launch_bounds__(128)
k_wmma_gemm(const _Float16* __restrict__ A,
            const _Float16* __restrict__ Bt,
            float* __restrict__ outF32,
            _Float16* __restrict__ outF16,
            const float* __restrict__ bias,
            int M, int Kpad, int Nvalid, int nTiles,
            int outStride, int applySigmoid) {
    __shared__ alignas(32) char sbuf[2][4096];

    const int tid  = threadIdx.x;
    const int lane = tid & 31;
    const int wv   = tid >> 5;
    const int mTiles  = (M + 15) >> 4;
    const int nGroups = (nTiles + 3) >> 2;
    const int mg = blockIdx.x / nGroups;
    const int ng = blockIdx.x - mg * nGroups;
    const int nBase = ng * 64;             // first N column of this block's slab
    const int mt    = mg * 4 + wv;         // this wave's m-tile (may exceed mTiles: guarded)
    const int half  = lane >> 4;
    const int l16   = lane & 15;
    const int npadRows = nTiles * 16;      // rows allocated in Bt

    int aRow = mt * 16 + l16;
    if (aRow >= M) aRow = M - 1;           // clamp; clamped rows never stored
    const _Float16* Ar = A + (size_t)aRow * Kpad;

    // async-copy source: thread t covers 32B of the 4KB slab:
    //   slab row (n within 64) = t>>1, byte cols = (t&1)*32 .. +32
    int grow = nBase + (tid >> 1);
    if (grow >= npadRows) grow = npadRows - 1;           // clamp (guarded on store)
    const char* gsrc = (const char*)(Bt + (size_t)grow * Kpad) + (tid & 1) * 32;
    const uint32 myLds0 = lds_off_(&sbuf[0][0]) + (uint32)(tid * 32);
    const uint32 myLds1 = lds_off_(&sbuf[1][0]) + (uint32)(tid * 32);

    v8f acc0 = {0.f,0.f,0.f,0.f,0.f,0.f,0.f,0.f};
    v8f acc1 = acc0, acc2 = acc0, acc3 = acc0;

    const int nk = Kpad >> 5;

    // prefetch k-tile 0 into buffer 0
    async_b128_(myLds0,      gsrc);
    async_b128_(myLds0 + 16, gsrc + 16);

    for (int kt = 0; kt < nk; ++kt) {
        const int kb = kt << 5;
        const char* sb = (kt & 1) ? &sbuf[1][0] : &sbuf[0][0];
        if (kt + 1 < nk) {
            const char* gs = gsrc + (size_t)(kt + 1) * 64;
            const uint32 l = (kt & 1) ? myLds0 : myLds1;   // next buffer
            async_b128_(l,      gs);
            async_b128_(l + 16, gs + 16);
            wait_async_<2>();                 // current buffer's 2 copies done
        } else {
            wait_async_<0>();
        }
        __syncthreads();                      // all waves' slab writes visible

        // A fragment (contiguous K to match B): VGPR v: K = half*16 + 2v+{0,1}
        const uint32* alo = (const uint32*)(Ar + kb + half * 16);
        const uint32* ahi = (const uint32*)(Ar + kb + half * 16 + 8);
        v8u au;
        au.s0 = alo[0]; au.s1 = alo[1]; au.s2 = alo[2]; au.s3 = alo[3];
        au.s4 = ahi[0]; au.s5 = ahi[1]; au.s6 = ahi[2]; au.s7 = ahi[3];
        v16h a = __builtin_bit_cast(v16h, au);

        // B fragments from LDS: n-tile j, lane = column nBase+j*16+l16,
        // contiguous 32B at row byte offset half*32 (K = half*16 + 0..15)
        v16h b0 = *(const v16h*)(sb + ((0 * 16 + l16) * 64 + half * 32));
        v16h b1 = *(const v16h*)(sb + ((1 * 16 + l16) * 64 + half * 32));
        v16h b2 = *(const v16h*)(sb + ((2 * 16 + l16) * 64 + half * 32));
        v16h b3 = *(const v16h*)(sb + ((3 * 16 + l16) * 64 + half * 32));

        acc0 = __builtin_amdgcn_wmma_f32_16x16x32_f16(false, a, false, b0, (short)0, acc0, false, false);
        acc1 = __builtin_amdgcn_wmma_f32_16x16x32_f16(false, a, false, b1, (short)0, acc1, false, false);
        acc2 = __builtin_amdgcn_wmma_f32_16x16x32_f16(false, a, false, b2, (short)0, acc2, false, false);
        acc3 = __builtin_amdgcn_wmma_f32_16x16x32_f16(false, a, false, b3, (short)0, acc3, false, false);

        __syncthreads();                      // slab reads done before it is overwritten
    }

    // store 4 tiles; C/D layout: VGPR r -> M = r + half*8, lane l16 -> N
#pragma unroll
    for (int j = 0; j < 4; ++j) {
        const v8f acc = (j == 0) ? acc0 : (j == 1) ? acc1 : (j == 2) ? acc2 : acc3;
        const int n = nBase + j * 16 + l16;
#pragma unroll
        for (int r = 0; r < 8; ++r) {
            const int m = mt * 16 + half * 8 + r;
            if (m >= M) continue;
            if (n < Nvalid) {
                float v = acc[r];
                if (bias) v += bias[n];
                if (applySigmoid) v = sigmoidf_(v);
                if (outF32) outF32[(size_t)m * outStride + n] = v;
                else        outF16[(size_t)m * outStride + n] = (_Float16)v;
            } else if (outF16 && n < nTiles * 16) {
                outF16[(size_t)m * outStride + n] = (_Float16)0.0f;   // keep pad clean
            }
        }
    }
}

// ---------------- SpMM scatter: S[row] += dinv[row]*dinv[col] * T[col] ----------------
__global__ void k_spmm(const int* __restrict__ er, const int* __restrict__ ec,
                       const float* __restrict__ dinv,
                       const float* __restrict__ T, float* __restrict__ S) {
    int e = blockIdx.x;
    int f = blockIdx.y * 32 + threadIdx.x;
    if (f >= EMB_) return;
    int r = er[e], c = ec[e];
    float v = dinv[r] * dinv[c];
    atomicAdd(&S[(size_t)r * EMBP_N + f], v * T[(size_t)c * EMBP_N + f]);
}

// sigmoid(S) -> Hf16 (zero K-pad)
__global__ void k_sig_cvt(const float* __restrict__ S, _Float16* __restrict__ Hf, int total) {
    int i = blockIdx.x * blockDim.x + threadIdx.x;
    if (i >= total) return;
    int f = i % EMBP_K;
    int v = i / EMBP_K;
    Hf[i] = (f < EMB_) ? (_Float16)sigmoidf_(S[(size_t)v * EMBP_N + f]) : (_Float16)0.0f;
}

// fc = concat(H2[node], input) -> FCf16 [BATCH][FCKP_], zero pad
__global__ void k_gather_concat(const _Float16* __restrict__ Hf,
                                const int* __restrict__ node,
                                const float* __restrict__ input,
                                _Float16* __restrict__ FC, int total) {
    int i = blockIdx.x * blockDim.x + threadIdx.x;
    if (i >= total) return;
    int f = i % FCKP_;
    int b = i / FCKP_;
    _Float16 v = (_Float16)0.0f;
    if (f < EMB_) {
        int r = node[b];
        v = Hf[(size_t)r * EMBP_K + f];
    } else if (f < FCK_) {
        v = (_Float16)input[(size_t)b * INDIM_ + (f - EMB_)];
    }
    FC[i] = v;
}

// out = h2 @ Wc + bc  (N=2: scalar dot per output element)
__global__ void k_final(const _Float16* __restrict__ H2,
                        const float* __restrict__ Wc, const float* __restrict__ bc,
                        float* __restrict__ out) {
    int t = blockIdx.x * blockDim.x + threadIdx.x;
    if (t >= BATCH_ * NLAB_) return;
    int i = t >> 1;
    int l = t & 1;
    float s = 0.0f;
    const _Float16* h = H2 + (size_t)i * HIDP_;
    for (int k = 0; k < HID_; ++k) s += (float)h[k] * Wc[k * NLAB_ + l];
    out[t] = s + bc[l];
}

// ---------------- launch ----------------

extern "C" void kernel_launch(void* const* d_in, const int* in_sizes, int n_in,
                              void* d_out, int out_size, void* d_ws, size_t ws_size,
                              hipStream_t stream) {
    const float* input = (const float*)d_in[0];
    const int*   node  = (const int*)  d_in[1];
    const int*   er    = (const int*)  d_in[2];
    const int*   ec    = (const int*)  d_in[3];
    const float* H0    = (const float*)d_in[4];
    const float* W1    = (const float*)d_in[5];
    const float* W2    = (const float*)d_in[6];
    const float* Wa    = (const float*)d_in[7];
    const float* ba    = (const float*)d_in[8];
    const float* Wb    = (const float*)d_in[9];
    const float* bb    = (const float*)d_in[10];
    const float* Wc    = (const float*)d_in[11];
    const float* bc    = (const float*)d_in[12];
    float* out = (float*)d_out;

    // workspace carve-out (256B aligned slots)
    char* ws = (char*)d_ws;
    size_t off = 0;
    auto carve = [&](size_t bytes) -> void* {
        void* p = ws + off;
        off = (off + bytes + 255) & ~(size_t)255;
        return p;
    };
    float*    deg  = (float*)   carve((size_t)NV_ * 4);
    float*    dinv = (float*)   carve((size_t)NV_ * 4);
    _Float16* Hf   = (_Float16*)carve((size_t)NV_ * EMBP_K * 2);
    float*    T    = (float*)   carve((size_t)NV_ * EMBP_N * 4);
    float*    S    = (float*)   carve((size_t)NV_ * EMBP_N * 4);
    _Float16* W1t  = (_Float16*)carve((size_t)EMBP_N * EMBP_K * 2);
    _Float16* W2t  = (_Float16*)carve((size_t)EMBP_N * EMBP_K * 2);
    _Float16* Wat  = (_Float16*)carve((size_t)HIDP_ * FCKP_ * 2);
    _Float16* Wbt  = (_Float16*)carve((size_t)HIDP_ * HIDP_ * 2);
    _Float16* FC   = (_Float16*)carve((size_t)BATCH_ * FCKP_ * 2);
    _Float16* Hh1  = (_Float16*)carve((size_t)BATCH_ * HIDP_ * 2);
    _Float16* Hh2  = (_Float16*)carve((size_t)BATCH_ * HIDP_ * 2);

    const int B256 = 256;
    auto blocks = [](long long n, int b) { return (int)((n + b - 1) / b); };
    auto gemmGrid = [](int M, int nTiles) {
        int mGroups = ((M + 15) / 16 + 3) / 4;
        int nGroups = (nTiles + 3) / 4;
        return mGroups * nGroups;
    };
    const int GEMM_BLK = 128;

    // 1) degree + dinv
    k_zero_f32<<<blocks(NV_, B256), B256, 0, stream>>>(deg, NV_);
    k_degree<<<blocks(NE_, B256), B256, 0, stream>>>(er, deg, NE_);
    k_dinv<<<blocks(NV_, B256), B256, 0, stream>>>(deg, dinv, NV_);

    // 2) precision conversions
    k_cvt_h0<<<blocks((long long)NV_ * EMBP_K, B256), B256, 0, stream>>>(H0, Hf, NV_ * EMBP_K);
    k_cvt_w<<<blocks((long long)EMBP_N * EMBP_K, B256), B256, 0, stream>>>(W1, W1t, EMB_, EMB_, EMBP_K, EMBP_N);
    k_cvt_w<<<blocks((long long)EMBP_N * EMBP_K, B256), B256, 0, stream>>>(W2, W2t, EMB_, EMB_, EMBP_K, EMBP_N);
    k_cvt_w<<<blocks((long long)HIDP_ * FCKP_, B256), B256, 0, stream>>>(Wa, Wat, FCK_, HID_, FCKP_, HIDP_);
    k_cvt_w<<<blocks((long long)HIDP_ * HIDP_, B256), B256, 0, stream>>>(Wb, Wbt, HID_, HID_, HIDP_, HIDP_);

    // 3) layer 1: T = H0f @ W1 ; S = LM*T ; H1 = sigmoid(S)
    k_wmma_gemm<<<gemmGrid(NV_, EMBP_N / 16), GEMM_BLK, 0, stream>>>(Hf, W1t, T, (_Float16*)nullptr,
        (const float*)nullptr, NV_, EMBP_K, EMB_, EMBP_N / 16, EMBP_N, 0);
    k_zero_f32<<<blocks((long long)NV_ * EMBP_N, B256), B256, 0, stream>>>(S, NV_ * EMBP_N);
    {
        dim3 g(NE_, (EMB_ + 31) / 32);
        k_spmm<<<g, 32, 0, stream>>>(er, ec, dinv, T, S);
    }
    k_sig_cvt<<<blocks((long long)NV_ * EMBP_K, B256), B256, 0, stream>>>(S, Hf, NV_ * EMBP_K);

    // 4) layer 2: T = H1f @ W2 ; S = LM*T ; H2 = sigmoid(S)
    k_wmma_gemm<<<gemmGrid(NV_, EMBP_N / 16), GEMM_BLK, 0, stream>>>(Hf, W2t, T, (_Float16*)nullptr,
        (const float*)nullptr, NV_, EMBP_K, EMB_, EMBP_N / 16, EMBP_N, 0);
    k_zero_f32<<<blocks((long long)NV_ * EMBP_N, B256), B256, 0, stream>>>(S, NV_ * EMBP_N);
    {
        dim3 g(NE_, (EMB_ + 31) / 32);
        k_spmm<<<g, 32, 0, stream>>>(er, ec, dinv, T, S);
    }
    k_sig_cvt<<<blocks((long long)NV_ * EMBP_K, B256), B256, 0, stream>>>(S, Hf, NV_ * EMBP_K);

    // 5) gather + concat
    k_gather_concat<<<blocks((long long)BATCH_ * FCKP_, B256), B256, 0, stream>>>(
        Hf, node, input, FC, BATCH_ * FCKP_);

    // 6) MLP: h1 = sigmoid(FC @ Wa + ba); h2 = sigmoid(h1 @ Wb + bb)
    k_wmma_gemm<<<gemmGrid(BATCH_, HIDP_ / 16), GEMM_BLK, 0, stream>>>(FC, Wat, (float*)nullptr, Hh1,
        ba, BATCH_, FCKP_, HID_, HIDP_ / 16, HIDP_, 1);
    k_wmma_gemm<<<gemmGrid(BATCH_, HIDP_ / 16), GEMM_BLK, 0, stream>>>(Hh1, Wbt, (float*)nullptr, Hh2,
        bb, BATCH_, HIDP_, HID_, HIDP_ / 16, HIDP_, 1);

    // 7) logits
    k_final<<<blocks((long long)BATCH_ * NLAB_, B256), B256, 0, stream>>>(Hh2, Wc, bc, out);
}
